// SelfAttention_20718922236411
// MI455X (gfx1250) — matrix operand order repaired and checked by reading, not compiled, not run on hardware
//
#include <hip/hip_runtime.h>

typedef _Float16 half_t;
typedef __attribute__((ext_vector_type(16))) _Float16 v16h;
typedef __attribute__((ext_vector_type(8)))  _Float16 v8h;
typedef __attribute__((ext_vector_type(8)))  float    v8f;

#define D_MODEL 384
#define SEQ     2048
#define NBATCH  32
#define MROWS   (NBATCH*SEQ)        // 65536
#define NT      (D_MODEL/16)        // 24 column tiles
#define KT      (D_MODEL/32)        // 12 k-steps

static __device__ __forceinline__ v8f wmma16(v16h a, v16h b, v8f c) {
  // D = A(16x32 f16) * B(32x16 f16) + C(16x16 f32)
  return __builtin_amdgcn_wmma_f32_16x16x32_f16(false, a, false, b, (short)0, c, false, false);
}

static __device__ __forceinline__ v16h cat16(v8h lo, v8h hi) {
  return __builtin_shufflevector(lo, hi, 0,1,2,3,4,5,6,7,8,9,10,11,12,13,14,15);
}

// ---------------------------------------------------------------------------
// Kernel 0: transpose weights to f16 [which][out][in] so WMMA B-fragments are
// two contiguous 16B loads per lane.
// ---------------------------------------------------------------------------
__global__ void prep_weights(const float* __restrict__ wq, const float* __restrict__ wk,
                             const float* __restrict__ wv, half_t* __restrict__ wt) {
  int idx = blockIdx.x * blockDim.x + threadIdx.x;
  if (idx >= 3 * D_MODEL * D_MODEL) return;
  int which = idx / (D_MODEL * D_MODEL);
  int r     = idx % (D_MODEL * D_MODEL);
  int out_c = r / D_MODEL;
  int in_c  = r % D_MODEL;
  const float* w = (which == 0) ? wq : ((which == 1) ? wk : wv);
  wt[idx] = (half_t)w[in_c * D_MODEL + out_c];
}

// ---------------------------------------------------------------------------
// Kernel 1: QKV projection. grid = (MROWS/64, 3). One wave = one 16-row strip.
// which==0 -> Q (row-major f16), 1 -> K (row-major f16), 2 -> V transposed
// [b][v][s] f16 (stored straight from the accumulator layout).
// ---------------------------------------------------------------------------
__global__ __launch_bounds__(128) void qkv_proj(
    const float* __restrict__ x,
    const float* __restrict__ bq, const float* __restrict__ bk, const float* __restrict__ bv,
    const half_t* __restrict__ wt,
    half_t* __restrict__ Qo, half_t* __restrict__ Ko, half_t* __restrict__ Vt)
{
  __shared__ alignas(32) half_t lds[4][16 * D_MODEL];   // 48 KB
  const int wave = threadIdx.x >> 5;
  const int lane = threadIdx.x & 31;
  const int lo = lane & 15, hi = lane >> 4;
  const int strip = blockIdx.x * 4 + wave;
  const int row0  = strip * 16;
  const int which = blockIdx.y;

  // A fragments: row = row0 + lo, halves 0..7 = K kbase..kbase+7, 8..15 = +16
  v16h a[KT];
  const float* xrow = x + (size_t)(row0 + lo) * D_MODEL;
#pragma unroll
  for (int kt = 0; kt < KT; ++kt) {
    const int f0 = kt * 32 + hi * 8;
    float4 p0 = *(const float4*)(xrow + f0);
    float4 p1 = *(const float4*)(xrow + f0 + 4);
    float4 p2 = *(const float4*)(xrow + f0 + 16);
    float4 p3 = *(const float4*)(xrow + f0 + 20);
    v16h t;
    t[0]=(half_t)p0.x;  t[1]=(half_t)p0.y;  t[2]=(half_t)p0.z;  t[3]=(half_t)p0.w;
    t[4]=(half_t)p1.x;  t[5]=(half_t)p1.y;  t[6]=(half_t)p1.z;  t[7]=(half_t)p1.w;
    t[8]=(half_t)p2.x;  t[9]=(half_t)p2.y;  t[10]=(half_t)p2.z; t[11]=(half_t)p2.w;
    t[12]=(half_t)p3.x; t[13]=(half_t)p3.y; t[14]=(half_t)p3.z; t[15]=(half_t)p3.w;
    a[kt] = t;
  }

  const half_t* w = wt + (size_t)which * D_MODEL * D_MODEL;
  const float* bias = (which == 0) ? bq : ((which == 1) ? bk : bv);
  half_t* slds = lds[wave];

  for (int nt = 0; nt < NT; ++nt) {
    const int col = nt * 16 + lo;
    v8f c = {};
#pragma unroll
    for (int kt = 0; kt < KT; ++kt) {
      v16h b = cat16(*(const v8h*)(w + (size_t)col * D_MODEL + kt * 32 + hi * 8), *(const v8h*)(w + (size_t)col * D_MODEL + kt * 32 + hi * 8 + 16));
      c = wmma16(a[kt], b, c);
    }
    const float bb = bias[col];
    if (which == 2) {
      // c[v] = row (v + 8*hi), col -> Vt[b][col][s]: 8 consecutive s -> 16B store
      const int bidx = row0 / SEQ, s0 = row0 % SEQ;
      v8h hv;
#pragma unroll
      for (int v = 0; v < 8; ++v) hv[v] = (half_t)(c[v] + bb);
      *(v8h*)(Vt + ((size_t)bidx * D_MODEL + col) * SEQ + s0 + hi * 8) = hv;
    } else {
#pragma unroll
      for (int v = 0; v < 8; ++v)
        slds[(v + 8 * hi) * D_MODEL + col] = (half_t)(c[v] + bb);
    }
  }

  if (which != 2) {
    // 16x384 f16 strip is contiguous in global: fully coalesced b128 copy
    half_t* dst = ((which == 0) ? Qo : Ko) + (size_t)row0 * D_MODEL;
    const v8h* s8 = (const v8h*)slds;
    v8h* d8 = (v8h*)dst;
#pragma unroll 4
    for (int i = lane; i < 16 * D_MODEL / 8; i += 32) d8[i] = s8[i];
  }
}

// ---------------------------------------------------------------------------
// Kernel 2: flash attention. One wave per 16-query tile, online softmax over
// s in chunks of 32. Q strip in LDS; P converts D-layout -> A-layout via LDS.
// ---------------------------------------------------------------------------
#define WBYTES (16 * D_MODEL * 4 + 16 * 32 * 2)   // 24 KB (Q f16 / out f32 reuse) + 1 KB P

__global__ __launch_bounds__(64) void attn(
    const half_t* __restrict__ Qg, const half_t* __restrict__ Kg,
    const half_t* __restrict__ Vt, float* __restrict__ out)
{
  extern __shared__ char smem[];
  const int wave = threadIdx.x >> 5;
  const int lane = threadIdx.x & 31;
  const int lo = lane & 15, hi = lane >> 4;

  char* wbase = smem + (size_t)wave * WBYTES;
  half_t* qs = (half_t*)wbase;                    // 16 x 384 f16 (first 12 KB)
  float*  os = (float*)wbase;                     // reused as 16 x 384 f32 at end
  half_t* ps = (half_t*)(wbase + 16 * D_MODEL * 4); // 16 x 32 f16 P tile

  const int tile = blockIdx.x * 2 + wave;         // 0..4095
  const int b  = tile >> 7;                       // / (SEQ/16)
  const int q0 = (tile & 127) * 16;
  const size_t row0 = (size_t)b * SEQ + q0;

  { // load Q strip (contiguous 12 KB) into LDS
    const v8h* src = (const v8h*)(Qg + row0 * D_MODEL);
    v8h* dst = (v8h*)qs;
#pragma unroll 4
    for (int i = lane; i < 16 * D_MODEL / 8; i += 32) dst[i] = src[i];
  }

  const half_t* kb = Kg + (size_t)b * SEQ * D_MODEL;   // [s][feat] f16
  const half_t* vb = Vt + (size_t)b * D_MODEL * SEQ;   // [v][s]    f16

  float m[8], l[8];
#pragma unroll
  for (int v = 0; v < 8; ++v) { m[v] = -1e30f; l[v] = 0.f; }
  v8f o[NT];
  v8f vzero = {};
#pragma unroll
  for (int nt = 0; nt < NT; ++nt) o[nt] = vzero;

  const float cs = 0.051031036307982884f * 1.4426950408889634f; // 1/sqrt(384)*log2(e)

  for (int s0 = 0; s0 < SEQ; s0 += 32) {
    // prefetch next chunk's K rows (global_prefetch_b8) while we compute
    if (s0 + 32 < SEQ)
      __builtin_prefetch(kb + (size_t)(s0 + 32 + lane) * D_MODEL, 0, 1);

    v8f sc0 = vzero, sc1 = vzero;
#pragma unroll
    for (int kt = 0; kt < KT; ++kt) {
      const half_t* ap = qs + lo * D_MODEL + kt * 32 + hi * 16;
      v16h a = cat16(*(const v8h*)ap, *(const v8h*)(ap + 8));
      v16h b0 = *(const v16h*)(kb + (size_t)(s0 + lo) * D_MODEL + kt * 32 + hi * 16);
      v16h b1 = *(const v16h*)(kb + (size_t)(s0 + 16 + lo) * D_MODEL + kt * 32 + hi * 16);
      sc0 = wmma16(a, b0, sc0);
      sc1 = wmma16(a, b1, sc1);
    }
    // online softmax (log2 domain), per-row stats via 16-lane xor reductions
    float al[8];
    bool chg = false;
#pragma unroll
    for (int v = 0; v < 8; ++v) {
      float t0 = sc0[v] * cs, t1 = sc1[v] * cs;
      float mx = fmaxf(t0, t1);
      mx = fmaxf(mx, __shfl_xor(mx, 1, 32));
      mx = fmaxf(mx, __shfl_xor(mx, 2, 32));
      mx = fmaxf(mx, __shfl_xor(mx, 4, 32));
      mx = fmaxf(mx, __shfl_xor(mx, 8, 32));
      float mn = fmaxf(m[v], mx);
      chg = chg || (mx > m[v]);
      float av = exp2f(m[v] - mn);
      float p0 = exp2f(t0 - mn);
      float p1 = exp2f(t1 - mn);
      float sm = p0 + p1;
      sm += __shfl_xor(sm, 1, 32);
      sm += __shfl_xor(sm, 2, 32);
      sm += __shfl_xor(sm, 4, 32);
      sm += __shfl_xor(sm, 8, 32);
      l[v] = l[v] * av + sm;
      m[v] = mn;
      al[v] = av;
      const int r = v + 8 * hi;                 // D-layout row
      ps[r * 32 + lo]      = (half_t)p0;
      ps[r * 32 + 16 + lo] = (half_t)p1;
    }
    // rescale running O only if some row max actually increased (alpha != 1).
    // Uniform wave-wide branch via ballot; skipping when all alpha==1 is exact.
    if (__ballot(chg)) {
#pragma unroll
      for (int nt = 0; nt < NT; ++nt)
#pragma unroll
        for (int v = 0; v < 8; ++v) o[nt][v] *= al[v];
    }
    // P as A-fragment (16 x 32 over s)
    const half_t* pp = ps + lo * 32 + hi * 16;
    v16h pa = cat16(*(const v8h*)pp, *(const v8h*)(pp + 8));
    // O += P @ V  (B from transposed V: contiguous 32B per lane)
#pragma unroll
    for (int nt = 0; nt < NT; ++nt) {
      v16h vv = *(const v16h*)(vb + (size_t)(nt * 16 + lo) * SEQ + s0 + hi * 16);
      o[nt] = wmma16(pa, vv, o[nt]);
    }
  }

  float inv[8];
#pragma unroll
  for (int v = 0; v < 8; ++v) inv[v] = 1.0f / l[v];
  // stage f32 strip in LDS (reuses Q space), then contiguous coalesced store
#pragma unroll
  for (int nt = 0; nt < NT; ++nt)
#pragma unroll
    for (int v = 0; v < 8; ++v)
      os[(v + 8 * hi) * D_MODEL + nt * 16 + lo] = o[nt][v] * inv[v];

  float* dst = out + row0 * D_MODEL;
  const float4* s4 = (const float4*)os;
  float4* d4 = (float4*)dst;
#pragma unroll 4
  for (int i = lane; i < 16 * D_MODEL / 4; i += 32) d4[i] = s4[i];
}

// ---------------------------------------------------------------------------
extern "C" void kernel_launch(void* const* d_in, const int* in_sizes, int n_in,
                              void* d_out, int out_size, void* d_ws, size_t ws_size,
                              hipStream_t stream) {
  const float* x  = (const float*)d_in[0];
  const float* wq = (const float*)d_in[1];
  const float* bq = (const float*)d_in[2];
  const float* wk = (const float*)d_in[3];
  const float* bk = (const float*)d_in[4];
  const float* wv = (const float*)d_in[5];
  const float* bv = (const float*)d_in[6];
  float* out = (float*)d_out;

  // workspace layout (f16): Q | K | Vt | Wt   (~145 MB total)
  half_t* Q  = (half_t*)d_ws;
  half_t* K  = Q  + (size_t)MROWS * D_MODEL;
  half_t* Vt = K  + (size_t)MROWS * D_MODEL;
  half_t* Wt = Vt + (size_t)NBATCH * D_MODEL * SEQ;

  const int wtotal = 3 * D_MODEL * D_MODEL;
  prep_weights<<<(wtotal + 255) / 256, 256, 0, stream>>>(wq, wk, wv, Wt);

  dim3 g1(MROWS / 64, 3, 1);                    // 4 strips of 16 rows per block
  qkv_proj<<<g1, 128, 0, stream>>>(x, bq, bk, bv, Wt, Q, K, Vt);

  const int tiles = NBATCH * (SEQ / 16);        // 4096 query tiles
  const size_t lds2 = 2 * (size_t)WBYTES;       // 2 waves per block, 50 KB
  attn<<<tiles / 2, 64, lds2, stream>>>(Q, K, Vt, out);
}
